// SchNet_hidden_60653528154558
// MI455X (gfx1250) — matrix operand order repaired and checked by reading, not compiled, not run on hardware
//
#include <hip/hip_runtime.h>
#include <hip/hip_bf16.h>
#include <cstdint>
#include <cstddef>

// ---------------- problem constants (from reference) ----------------
#define H       128
#define NNODE   50000
#define EEDGE   800000
#define NGRAPHS 1000
#define NPG     50
#define DEG     16
#define NG      50
#define KG      64          // gaussians padded to 64 for K%32==0

typedef __attribute__((ext_vector_type(16))) __bf16        v16bf;
typedef __attribute__((ext_vector_type(8)))  float         v8f;
typedef __attribute__((ext_vector_type(4)))  unsigned int  v4u;

union BfPack {
  v4u            q[2];   // two 16B chunks
  v16bf          v;
  unsigned short s[16];
};

__device__ __forceinline__ unsigned short f2bf(float f) {
  unsigned int x = __float_as_uint(f);
  x += 0x7FFFu + ((x >> 16) & 1u);           // round-to-nearest-even
  return (unsigned short)(x >> 16);
}

__device__ __forceinline__ v8f vzero8() {
  v8f z;
#pragma unroll
  for (int i = 0; i < 8; ++i) z[i] = 0.0f;
  return z;
}

// scheduling fence: keep the batched loads from being sunk into the WMMA chain
__device__ __forceinline__ void sched_fence() {
#if __has_builtin(__builtin_amdgcn_sched_barrier)
  __builtin_amdgcn_sched_barrier(0);
#endif
}

__device__ __forceinline__ v8f wmma_bf16(v16bf a, v16bf b, v8f c) {
  // D = A(16x32 bf16) * B(32x16 bf16) + C(16x16 f32)
  return __builtin_amdgcn_wmma_f32_16x16x32_bf16(false, a, false, b, (short)0, c,
                                                 false, false);
}

// B operand: packed tiles of 512 bf16, lane-major (lane holds one column,
// K 8h..8h+7 and 16+8h..+7 in lane half h) -> two contiguous b128 loads per lane.
__device__ __forceinline__ void load_b_tile(const unsigned short* __restrict__ bp,
                                            int tile, int lane, BfPack& b) {
  const v4u* p = (const v4u*)(bp + (size_t)tile * 512 + lane * 16);
  b.q[0] = p[0];
  b.q[1] = p[1];
}

// A operand from row-major bf16: lane(lo,lhi) -> row, K chunks
// {kb+8*lhi .. +7} and {kb+16+8*lhi .. +7}  (ISA 16-bit A 16x32 layout)
__device__ __forceinline__ void load_a_rowmajor(const unsigned short* __restrict__ A,
                                                size_t row, int lda, int kb, int lhi,
                                                BfPack& a) {
  const unsigned short* r = A + row * (size_t)lda + kb + 8 * lhi;
  a.q[0] = *(const v4u*)(r);
  a.q[1] = *(const v4u*)(r + 16);
}

// one k-step: batch-load all 8 B tiles, fence, then 8 back-to-back WMMAs.
// The fence keeps all 8 tiles live so waits are staged instead of wait(0).
__device__ __forceinline__ void kstep_mma(const BfPack& a,
                                          const unsigned short* __restrict__ Bp,
                                          int ks, int lane, v8f acc[8]) {
  BfPack bt[8];
#pragma unroll
  for (int nt = 0; nt < 8; ++nt) load_b_tile(Bp, ks * 8 + nt, lane, bt[nt]);
  sched_fence();
#pragma unroll
  for (int nt = 0; nt < 8; ++nt) acc[nt] = wmma_bf16(a.v, bt[nt].v, acc[nt]);
}

// ---------------- weight repack: fp32 [K,H] row-major -> WMMA B-layout bf16 --
__global__ void pack_b(const float* __restrict__ src, unsigned short* __restrict__ dst,
                       int Kpad, int Ksrc) {
  int id = blockIdx.x * blockDim.x + threadIdx.x;
  int total = Kpad * 128;                 // (Kpad/32)*8 tiles * 512 elems
  if (id >= total) return;
  int tile = id >> 9;                     // kstep*8 + ntile
  int rem  = id & 511;
  int lane = rem >> 4;
  int j    = rem & 15;
  int kstep = tile >> 3;
  int ntile = tile & 7;
  int n = ntile * 16 + (lane & 15);
  int K = kstep * 32 + (j < 8 ? (lane >> 4) * 8 + j : 16 + (lane >> 4) * 8 + (j - 8));
  float v = (K < Ksrc) ? src[(size_t)K * H + n] : 0.0f;
  dst[id] = f2bf(v);
}

// ---------------- elementwise fp32 -> bf16 (embedding tables) ---------------
__global__ void cvt_f32_bf16(const float* __restrict__ src,
                             unsigned short* __restrict__ dst, int n) {
  int i = blockIdx.x * blockDim.x + threadIdx.x;
  if (i < n) dst[i] = f2bf(src[i]);
}

// ---------------- node embedding + lin1 (A gathered from bf16 tables) -------
__global__ void __launch_bounds__(128, 1)
embed_lin1(const int* __restrict__ z,
           const unsigned short* __restrict__ tb1,
           const unsigned short* __restrict__ tb2,
           const unsigned short* __restrict__ tb3,
           const unsigned short* __restrict__ tb4,
           const unsigned short* __restrict__ Bp,
           const float* __restrict__ bias,
           float* __restrict__ v, unsigned short* __restrict__ vbf,
           int ntiles) {
  int wv = blockIdx.x * (blockDim.x >> 5) + (threadIdx.x >> 5);
  if (wv >= ntiles) return;
  int lane = threadIdx.x & 31, lo = lane & 15, lhi = lane >> 4;
  int node = wv * 16 + lo;
  const unsigned short* tabs[4] = {tb1, tb2, tb3, tb4};
  int idx4[4];
#pragma unroll
  for (int j = 0; j < 4; ++j) idx4[j] = z[(size_t)node * 4 + j];

  v8f acc[8];
#pragma unroll
  for (int i = 0; i < 8; ++i) acc[i] = vzero8();

#pragma unroll
  for (int ks = 0; ks < 16; ++ks) {           // K = 512 = 16 * 32
    int j   = ks >> 2;                        // which embedding table
    int klb = (ks & 3) * 32;                  // local K base inside table
    BfPack a;
    load_a_rowmajor(tabs[j], (size_t)idx4[j], H, klb, lhi, a);
    kstep_mma(a, Bp, ks, lane, acc);
  }
#pragma unroll
  for (int nt = 0; nt < 8; ++nt) {
    float bv = bias[nt * 16 + lo];
#pragma unroll
    for (int i = 0; i < 8; ++i) {             // C layout: m = i + 8*lhi, n = lo
      int m = i + 8 * lhi;
      size_t cidx = ((size_t)wv * 16 + m) * H + nt * 16 + lo;
      float val = acc[nt][i] + bv;
      v[cidx]   = val;
      vbf[cidx] = f2bf(val);
    }
  }
}

// ---------------- edge preprocessing: distance, RBF (bf16, K padded), cutoff -
__global__ void edge_prep(const int* __restrict__ ei, const float* __restrict__ pos,
                          unsigned short* __restrict__ demb, float* __restrict__ cc) {
  int e = blockIdx.x * blockDim.x + threadIdx.x;
  if (e >= EEDGE) return;
  int r = ei[e], c = ei[EEDGE + e];
  float dx = pos[r * 3 + 0] - pos[c * 3 + 0];
  float dy = pos[r * 3 + 1] - pos[c * 3 + 1];
  float dz = pos[r * 3 + 2] - pos[c * 3 + 2];
  float d  = sqrtf(dx * dx + dy * dy + dz * dz);
  cc[e] = 0.5f * (__cosf(d * 0.6283185307179586f) + 1.0f);   // pi/5
  const float step  = 5.0f / 49.0f;
  const float coeff = -0.5f / (step * step);
  alignas(16) unsigned short o[KG];
#pragma unroll
  for (int g = 0; g < KG; ++g) {
    float t = d - (float)g * step;
    float val = (g < NG) ? __expf(coeff * t * t) : 0.0f;
    o[g] = f2bf(val);
  }
  v4u* dp = (v4u*)(demb + (size_t)e * KG);
  const v4u* sp = (const v4u*)o;
#pragma unroll
  for (int q = 0; q < 8; ++q) dp[q] = sp[q];
}

// ---------------- generic 16-row-tile GEMM: D = [relu](A_bf16 @ Bpack + bias) -
template <int KS>
__global__ void __launch_bounds__(128, 1)
gemm_tile16(const unsigned short* __restrict__ A, int lda,
            const unsigned short* __restrict__ Bp,
            const float* __restrict__ bias, int do_relu,
            float* __restrict__ D, int ntiles, int maxrow) {
  int wv = blockIdx.x * (blockDim.x >> 5) + (threadIdx.x >> 5);
  if (wv >= ntiles) return;
  int lane = threadIdx.x & 31, lo = lane & 15, lhi = lane >> 4;
  size_t row = (size_t)wv * 16 + lo;
  v8f acc[8];
#pragma unroll
  for (int i = 0; i < 8; ++i) acc[i] = vzero8();

  if (KS <= 4) {
    // preload all A operands, then stream B tiles
    BfPack a[KS > 0 ? KS : 1];
#pragma unroll
    for (int ks = 0; ks < KS; ++ks) load_a_rowmajor(A, row, lda, ks * 32, lhi, a[ks]);
    sched_fence();
#pragma unroll
    for (int ks = 0; ks < KS; ++ks) kstep_mma(a[ks], Bp, ks, lane, acc);
  } else {
#pragma unroll
    for (int ks = 0; ks < KS; ++ks) {
      BfPack a;
      load_a_rowmajor(A, row, lda, ks * 32, lhi, a);
      kstep_mma(a, Bp, ks, lane, acc);
    }
  }
#pragma unroll
  for (int nt = 0; nt < 8; ++nt) {
    float bv = bias ? bias[nt * 16 + lo] : 0.0f;
#pragma unroll
    for (int i = 0; i < 8; ++i) {
      int m = i + 8 * lhi;
      int r = wv * 16 + m;
      if (r < maxrow) {
        float val = acc[nt][i] + bv;
        if (do_relu) val = fmaxf(val, 0.0f);
        D[(size_t)r * H + nt * 16 + lo] = val;
      }
    }
  }
}

// ---------------- fused edge filter + gather + segment-sum ------------------
// One wave per destination node; its 16 contiguous edges form the WMMA M-tile.
// out[node] = sum_m  vl[row[m]] * ((relu(demb@W1+b1)@W2 + b2) * C[m])
__global__ void __launch_bounds__(128, 1)
edge_filter_aggr(const unsigned short* __restrict__ demb,
                 const unsigned short* __restrict__ W1p,
                 const float* __restrict__ b1,
                 const unsigned short* __restrict__ W2p,
                 const float* __restrict__ b2,
                 const float* __restrict__ cc,
                 const int* __restrict__ erow,
                 const float* __restrict__ vl,
                 unsigned short* __restrict__ outbf) {
  __shared__ alignas(16) unsigned short hid[4][16 * H];
  __shared__ int   rsh[4][16];
  __shared__ float csh[4][16];
  int w = threadIdx.x >> 5;
  int node = blockIdx.x * 4 + w;
  if (node >= NNODE) return;
  int lane = threadIdx.x & 31, lo = lane & 15, lhi = lane >> 4;
  size_t e0 = (size_t)node * DEG;
  if (lane < 16) {
    rsh[w][lane] = erow[e0 + lane];
    csh[w][lane] = cc[e0 + lane];
  }

  // GEMM1: hidden = relu(dist_emb @ W1 + b1), K = 64 (A preloaded: 2 k-steps)
  v8f acc[8];
#pragma unroll
  for (int i = 0; i < 8; ++i) acc[i] = vzero8();
  {
    BfPack a0, a1;
    load_a_rowmajor(demb, e0 + lo, KG, 0, lhi, a0);
    load_a_rowmajor(demb, e0 + lo, KG, 32, lhi, a1);
    sched_fence();
    kstep_mma(a0, W1p, 0, lane, acc);
    kstep_mma(a1, W1p, 1, lane, acc);
  }
  // C-layout -> row-major bf16 tile in LDS (transpose for GEMM2 A operand)
#pragma unroll
  for (int nt = 0; nt < 8; ++nt) {
    float bv = b1[nt * 16 + lo];
#pragma unroll
    for (int i = 0; i < 8; ++i) {
      int m = i + 8 * lhi;
      hid[w][m * H + nt * 16 + lo] = f2bf(fmaxf(acc[nt][i] + bv, 0.0f));
    }
  }
  __syncthreads();

  // GEMM2: Wf_pre = hidden @ W2, K = 128 (A preloaded from LDS: 4 k-steps)
#pragma unroll
  for (int i = 0; i < 8; ++i) acc[i] = vzero8();
  {
    BfPack a[4];
#pragma unroll
    for (int ks = 0; ks < 4; ++ks) {
      const unsigned short* hr = &hid[w][lo * H + ks * 32 + 8 * lhi];
      a[ks].q[0] = *(const v4u*)hr;
      a[ks].q[1] = *(const v4u*)(hr + 16);
    }
    sched_fence();
#pragma unroll
    for (int ks = 0; ks < 4; ++ks) kstep_mma(a[ks], W2p, ks, lane, acc);
  }

  // modulate by cutoff, gather vl[row], reduce the 16 edge rows -> 1 node row
  float sums[8];
#pragma unroll
  for (int nt = 0; nt < 8; ++nt) {
    float bv = b2[nt * 16 + lo];
    float part = 0.0f;
#pragma unroll
    for (int i = 0; i < 8; ++i) {
      int m = i + 8 * lhi;
      float wf = (acc[nt][i] + bv) * csh[w][m];
      part += wf * vl[(size_t)rsh[w][m] * H + nt * 16 + lo];
    }
    part += __shfl_xor(part, 16, 32);   // combine m-halves (lanes l <-> l+16)
    sums[nt] = part;
  }
  if (lane < 16) {
#pragma unroll
    for (int nt = 0; nt < 8; ++nt)
      outbf[(size_t)node * H + nt * 16 + lane] = f2bf(sums[nt]);
  }
}

// ---------------- node MLP + residual: v += relu(out@v1+b)@v2+b; xs slot -----
__global__ void __launch_bounds__(128, 1)
node_mlp_update(const unsigned short* __restrict__ outbf,
                const unsigned short* __restrict__ W1p,
                const float* __restrict__ b1,
                const unsigned short* __restrict__ W2p,
                const float* __restrict__ b2,
                float* __restrict__ v, unsigned short* __restrict__ vbf,
                unsigned short* __restrict__ xs, int layer, int ntiles) {
  __shared__ alignas(16) unsigned short hid[4][16 * H];
  int w = threadIdx.x >> 5;
  int wv = blockIdx.x * 4 + w;
  if (wv >= ntiles) return;
  int lane = threadIdx.x & 31, lo = lane & 15, lhi = lane >> 4;
  size_t row = (size_t)wv * 16 + lo;

  v8f acc[8];
#pragma unroll
  for (int i = 0; i < 8; ++i) acc[i] = vzero8();
  {
    BfPack a[4];
#pragma unroll
    for (int ks = 0; ks < 4; ++ks) load_a_rowmajor(outbf, row, H, ks * 32, lhi, a[ks]);
    sched_fence();
#pragma unroll
    for (int ks = 0; ks < 4; ++ks) kstep_mma(a[ks], W1p, ks, lane, acc);
  }
#pragma unroll
  for (int nt = 0; nt < 8; ++nt) {
    float bv = b1[nt * 16 + lo];
#pragma unroll
    for (int i = 0; i < 8; ++i) {
      int m = i + 8 * lhi;
      hid[w][m * H + nt * 16 + lo] = f2bf(fmaxf(acc[nt][i] + bv, 0.0f));
    }
  }
  __syncthreads();

#pragma unroll
  for (int i = 0; i < 8; ++i) acc[i] = vzero8();
  {
    BfPack a[4];
#pragma unroll
    for (int ks = 0; ks < 4; ++ks) {
      const unsigned short* hr = &hid[w][lo * H + ks * 32 + 8 * lhi];
      a[ks].q[0] = *(const v4u*)hr;
      a[ks].q[1] = *(const v4u*)(hr + 16);
    }
    sched_fence();
#pragma unroll
    for (int ks = 0; ks < 4; ++ks) kstep_mma(a[ks], W2p, ks, lane, acc);
  }
#pragma unroll
  for (int nt = 0; nt < 8; ++nt) {
    float bv = b2[nt * 16 + lo];
#pragma unroll
    for (int i = 0; i < 8; ++i) {
      int m = i + 8 * lhi;
      size_t node = (size_t)wv * 16 + m;
      size_t cidx = node * H + nt * 16 + lo;
      float nv = v[cidx] + acc[nt][i] + bv;
      v[cidx] = nv;
      unsigned short bf = f2bf(nv);
      vbf[cidx] = bf;
      xs[node * (3 * H) + (size_t)layer * H + nt * 16 + lo] = bf;
    }
  }
}

// ---------------- graph pooling: segment_sum over 50 contiguous nodes -------
__global__ void graph_pool(const float* __restrict__ h, unsigned short* __restrict__ upbf) {
  int g = blockIdx.x, t = threadIdx.x;      // grid 1008 (16-padded), 128 threads
  float s = 0.0f;
  if (g < NGRAPHS) {
    for (int i = 0; i < NPG; ++i) s += h[((size_t)g * NPG + i) * H + t];
  }
  upbf[(size_t)g * H + t] = f2bf(s);
}

// =============================== launcher ====================================
extern "C" void kernel_launch(void* const* d_in, const int* in_sizes, int n_in,
                              void* d_out, int out_size, void* d_ws, size_t ws_size,
                              hipStream_t stream) {
  (void)in_sizes; (void)n_in; (void)out_size; (void)ws_size;

  const int*   z    = (const int*)d_in[0];
  const float* pos  = (const float*)d_in[1];
  /* d_in[2] = batch: unused (graphs are 50 contiguous nodes) */
  const int*   ei   = (const int*)d_in[3];
  const float* emb1 = (const float*)d_in[4];
  const float* emb2 = (const float*)d_in[5];
  const float* emb3 = (const float*)d_in[6];
  const float* emb4 = (const float*)d_in[7];
  const float* lin1_w = (const float*)d_in[8];
  const float* lin1_b = (const float*)d_in[9];
  const float *mw1[3], *mb1[3], *mw2[3], *mb2[3], *linw[3], *v1w[3], *v1b[3], *v2w[3], *v2b[3];
  for (int l = 0; l < 3; ++l) {
    int b = 10 + l * 9;
    mw1[l]  = (const float*)d_in[b + 0];
    mb1[l]  = (const float*)d_in[b + 1];
    mw2[l]  = (const float*)d_in[b + 2];
    mb2[l]  = (const float*)d_in[b + 3];
    linw[l] = (const float*)d_in[b + 4];
    v1w[l]  = (const float*)d_in[b + 5];
    v1b[l]  = (const float*)d_in[b + 6];
    v2w[l]  = (const float*)d_in[b + 7];
    v2b[l]  = (const float*)d_in[b + 8];
  }
  const float* u1w = (const float*)d_in[37];
  const float* u1b = (const float*)d_in[38];
  const float* u2w = (const float*)d_in[39];
  const float* u2b = (const float*)d_in[40];

  // -------- workspace carve-up --------
  char* ws = (char*)d_ws;
  size_t off = 0;
  auto carve = [&](size_t bytes) -> char* {
    char* p = ws + off;
    off = (off + bytes + 255) & ~(size_t)255;
    return p;
  };
  float*          v_f   = (float*)         carve((size_t)NNODE * H * 4);
  unsigned short* vbf   = (unsigned short*)carve((size_t)NNODE * H * 2);
  float*          vl    = (float*)         carve((size_t)NNODE * H * 4);
  unsigned short* outbf = (unsigned short*)carve((size_t)NNODE * H * 2);
  unsigned short* xs    = (unsigned short*)carve((size_t)NNODE * 3 * H * 2);
  float*          hbuf  = (float*)         carve((size_t)NNODE * H * 4);
  unsigned short* upbf  = (unsigned short*)carve((size_t)1008 * H * 2);
  unsigned short* demb  = (unsigned short*)carve((size_t)EEDGE * KG * 2);
  float*          ccb   = (float*)         carve((size_t)EEDGE * 4);
  unsigned short* pk    = (unsigned short*)carve((size_t)352256 * 2);
  unsigned short* ebt   = (unsigned short*)carve((size_t)130 * H * 2); // bf16 tables

  unsigned short* tb1 = ebt;                 // 100 rows
  unsigned short* tb2 = ebt + 100 * H;       // 10 rows
  unsigned short* tb3 = ebt + 110 * H;
  unsigned short* tb4 = ebt + 120 * H;

  // packed-weight offsets (bf16 elements): tile = 512 elems = 32x16 B operand
  const size_t P_LIN1 = 0;                  // K=512 -> 128 tiles
  const size_t P_LYR  = 65536;              // per-layer block, stride 73728
  const size_t P_W1 = 0, P_W2 = 8192, P_LIN = 24576, P_V1 = 40960, P_V2 = 57344;
  const size_t P_U1 = 65536 + 3 * 73728;    // 286720, K=384 -> 96 tiles
  const size_t P_U2 = P_U1 + 49152;         // 335872

  auto pack = [&](const float* src, size_t dstoff, int Kpad, int Ksrc) {
    int total = Kpad * 128;
    pack_b<<<(total + 255) / 256, 256, 0, stream>>>(src, pk + dstoff, Kpad, Ksrc);
  };
  pack(lin1_w, P_LIN1, 512, 512);
  for (int l = 0; l < 3; ++l) {
    size_t b = P_LYR + (size_t)l * 73728;
    pack(mw1[l],  b + P_W1,  64, 50);       // 50 gaussians zero-padded to 64
    pack(mw2[l],  b + P_W2,  128, 128);
    pack(linw[l], b + P_LIN, 128, 128);
    pack(v1w[l],  b + P_V1,  128, 128);
    pack(v2w[l],  b + P_V2,  128, 128);
  }
  pack(u1w, P_U1, 384, 384);
  pack(u2w, P_U2, 128, 128);

  // embedding tables -> bf16 row-major
  cvt_f32_bf16<<<(100 * H + 255) / 256, 256, 0, stream>>>(emb1, tb1, 100 * H);
  cvt_f32_bf16<<<(10 * H + 255) / 256, 256, 0, stream>>>(emb2, tb2, 10 * H);
  cvt_f32_bf16<<<(10 * H + 255) / 256, 256, 0, stream>>>(emb3, tb3, 10 * H);
  cvt_f32_bf16<<<(10 * H + 255) / 256, 256, 0, stream>>>(emb4, tb4, 10 * H);

  const int tilesN = NNODE / 16;            // 3125
  const int gN     = (tilesN + 3) / 4;      // 4 waves / 128-thread block

  // node embedding + lin1  ->  v (f32) + vbf (bf16 A-operand copy)
  embed_lin1<<<gN, 128, 0, stream>>>(z, tb1, tb2, tb3, tb4,
                                     pk + P_LIN1, lin1_b, v_f, vbf, tilesN);
  // distances -> RBF (bf16 [E,64]) + cosine cutoff
  edge_prep<<<(EEDGE + 255) / 256, 256, 0, stream>>>(ei, pos, demb, ccb);

  for (int l = 0; l < 3; ++l) {
    size_t b = P_LYR + (size_t)l * 73728;
    // vl = v @ lin_w
    gemm_tile16<4><<<gN, 128, 0, stream>>>(vbf, H, pk + b + P_LIN,
                                           nullptr, 0, vl, tilesN, NNODE);
    // fused filter MLP + cutoff + gather + per-node segment sum
    edge_filter_aggr<<<(NNODE + 3) / 4, 128, 0, stream>>>(
        demb, pk + b + P_W1, mb1[l], pk + b + P_W2, mb2[l],
        ccb, ei, vl, outbf);
    // v += relu(out@v1+b1)@v2+b2 ; write xs[:, l*H : (l+1)*H]
    node_mlp_update<<<gN, 128, 0, stream>>>(
        outbf, pk + b + P_V1, v1b[l], pk + b + P_V2, v2b[l],
        v_f, vbf, xs, l, tilesN);
  }

  // h = relu(xs @ u1_w + u1_b)
  gemm_tile16<12><<<gN, 128, 0, stream>>>(xs, 3 * H, pk + P_U1,
                                          u1b, 1, hbuf, tilesN, NNODE);
  // graph pooling (50 contiguous nodes per graph), 16-padded rows zeroed
  graph_pool<<<1008, 128, 0, stream>>>(hbuf, upbf);
  // u = relu(upre @ u2_w + u2_b) -> d_out [1000,128] f32
  const int tilesG = 1008 / 16;             // 63
  gemm_tile16<4><<<(tilesG + 3) / 4, 128, 0, stream>>>(upbf, H, pk + P_U2,
                                                       u2b, 1, (float*)d_out,
                                                       tilesG, NGRAPHS);
}
